// PalBertSelfAttention_3066606649855
// MI455X (gfx1250) — matrix operand order repaired and checked by reading, not compiled, not run on hardware
//
#include <hip/hip_runtime.h>

// ---- CDNA5 WMMA vector types ----
typedef __bf16  v16bf __attribute__((ext_vector_type(16)));
typedef __bf16  v8bf  __attribute__((ext_vector_type(8)));
typedef __bf16  v4bf  __attribute__((ext_vector_type(4)));
typedef float   v8f   __attribute__((ext_vector_type(8)));
typedef unsigned v4u  __attribute__((ext_vector_type(4)));
typedef unsigned v8u  __attribute__((ext_vector_type(8)));

// Problem constants (match reference setup_inputs)
#define BATCH 4
#define SEQ   2048
#define DMODEL 1024
#define NHEAD 16
#define DHEAD 64
#define ROWS  (BATCH*SEQ)        // 8192
#define NQKV  (3*DMODEL)         // 3072

// ---------------------------------------------------------------------------
// CDNA5 data-mover helpers (see CDNA5 ISA)
// LDS operand: low 32 bits of the flat shared-aperture address == LDS offset.
// ---------------------------------------------------------------------------
__device__ __forceinline__ unsigned ldsoff(const void* p) {
    return (unsigned)(unsigned long long)p;
}
// ASYNCcnt-tracked per-lane async copy (global -> LDS), b128 per lane.
__device__ __forceinline__ void async_ld16(void* lds, const void* gmem) {
    unsigned loff = ldsoff(lds);
    asm volatile("global_load_async_to_lds_b128 %0, %1, off"
                 :: "v"(loff), "v"(gmem) : "memory");
}
__device__ __forceinline__ void wait_async_le4() {
    asm volatile("s_wait_asynccnt 4" ::: "memory");
}
__device__ __forceinline__ void wait_async_0() {
    asm volatile("s_wait_asynccnt 0" ::: "memory");
}

// TENSORcnt-tracked TDM tile load: 2D tile [tile_rows x 64] bf16 from a
// row-major tensor (row stride = stride_elems), LDS rows padded by 16B so the
// LDS row stride is 72 elements (bank spread). D# built per ISA 8.3/8.4:
//   pad_interval code 4 = 32 DWORDs (=128B row), pad_amount code 3 = 4 DWORDs.
__device__ __forceinline__ void tdm_load_tile64(unsigned lds_off, const void* gptr,
                                                unsigned tile_rows,
                                                unsigned stride_elems) {
    const unsigned long long ga = (unsigned long long)gptr;
    v4u g0;
    g0[0] = 1u;                                     // count=1, user descriptor
    g0[1] = lds_off;                                // lds_addr
    g0[2] = (unsigned)ga;                           // global_addr[31:0]
    g0[3] = ((unsigned)(ga >> 32) & 0x1FFFFFFu)     // global_addr[56:32]
          | 0x80000000u;                            // type=2 ("image")
    v8u g1;
    g1[0] = (1u << 16)                              // data_size = 2 bytes
          | (1u << 20)                              // pad_enable
          | (4u << 22)                              // pad_interval: 32 DWORDs
          | (3u << 25);                             // pad_amount: 4 DWORDs (16B)
    g1[1] = 0u;                                     // no atomic barrier; td0 lo16=0
    g1[2] = 0x10u;                                  // tensor_dim0 = 1M (hi16)
    g1[3] = 0x10u | (64u << 16);                    // tensor_dim1 = 1M; tile_dim0=64
    g1[4] = tile_rows;                              // tile_dim1; tile_dim2=0
    g1[5] = stride_elems;                           // tensor_dim0_stride[31:0]
    g1[6] = 0u;
    g1[7] = 0u;
    v4u gz; gz[0]=0u; gz[1]=0u; gz[2]=0u; gz[3]=0u; // groups 2/3 unused (2D)
    asm volatile("tensor_load_to_lds %0, %1, %2, %3"
                 :: "s"(g0), "s"(g1), "s"(gz), "s"(gz) : "memory");
}

__device__ __forceinline__ v16bf pack16(v8bf lo, v8bf hi) {
    v16bf r;
#pragma unroll
    for (int j = 0; j < 8; ++j) { r[j] = lo[j]; r[8+j] = hi[j]; }
    return r;
}

// ---------------------------------------------------------------------------
// Kernel 0: fp32 -> bf16 convert. hidden -> Xb [8192,1024];
// Wq|Wk|Wv -> Wcat [3072,1024].
// ---------------------------------------------------------------------------
__global__ __launch_bounds__(256) void cvt_kernel(
    const float* __restrict__ hidden,
    const float* __restrict__ Wq, const float* __restrict__ Wk,
    const float* __restrict__ Wv,
    __bf16* __restrict__ Xb, __bf16* __restrict__ Wcat)
{
    const long NX4 = (long)ROWS * DMODEL / 4;
    const long NW4 = (long)NQKV * DMODEL / 4;
    const long i4  = (long)blockIdx.x * blockDim.x + threadIdx.x;
    if (i4 < NX4) {
        float4 v = ((const float4*)hidden)[i4];
        v4bf o; o[0]=(__bf16)v.x; o[1]=(__bf16)v.y; o[2]=(__bf16)v.z; o[3]=(__bf16)v.w;
        *(v4bf*)(Xb + i4*4) = o;
    } else if (i4 < NX4 + NW4) {
        const long j = (i4 - NX4) * 4;
        const long n = j >> 10;
        const long k = j & (DMODEL - 1);
        const float* src = (n < DMODEL)   ? (Wq + n*DMODEL + k)
                         : (n < 2*DMODEL) ? (Wk + (n-DMODEL)*DMODEL + k)
                                          : (Wv + (n-2*DMODEL)*DMODEL + k);
        float4 v = *(const float4*)src;
        v4bf o; o[0]=(__bf16)v.x; o[1]=(__bf16)v.y; o[2]=(__bf16)v.z; o[3]=(__bf16)v.w;
        *(v4bf*)(Wcat + j) = o;
    }
}

// ---------------------------------------------------------------------------
// Kernel 1: fused QKV projection, TDM-staged, double buffered.
// Block = 256 thr (8 waves) computes 128(M) x 128(N); wave = 64(M) x 32(N).
// K-step 64: wave 0 issues two tensor_load_to_lds (A[128x64], B[128x64] bf16,
// LDS rows padded to 72 elems by the TDM), s_wait_tensorcnt + barrier, then
// 16 WMMAs per wave per step. Epilogue: Q,K -> [B,H,S,Dh]; V -> [B,H,Dh,S].
// ---------------------------------------------------------------------------
__global__ __launch_bounds__(256) void qkv_gemm_kernel(
    const __bf16* __restrict__ Xb,    // [8192][1024]
    const __bf16* __restrict__ Wcat,  // [3072][1024]
    const float* __restrict__ bq, const float* __restrict__ bk,
    const float* __restrict__ bv,
    __bf16* __restrict__ Qh, __bf16* __restrict__ Kh, __bf16* __restrict__ Vt)
{
    __shared__ __align__(16) __bf16 As[2][128][72];
    __shared__ __align__(16) __bf16 Bs[2][128][72];

    const int tid  = threadIdx.x;
    const int lane = tid & 31;
    const int wave = tid >> 5;
    const int half = lane >> 4;
    const int lc   = lane & 15;
    const int n_blk = blockIdx.x * 128;   // 24 blocks over N=3072
    const int m_blk = blockIdx.y * 128;   // 64 blocks over M=8192
    const int wm = (wave & 1) * 64;       // wave: 4 M-tiles
    const int wn = (wave >> 1) * 32;      // wave: 2 N-tiles

    auto issue = [&](int buf, int kb) {   // wave 0 only: 2 TDM descriptors
        tdm_load_tile64(ldsoff(&As[buf][0][0]),
                        Xb   + (long)m_blk*DMODEL + kb, 128u, (unsigned)DMODEL);
        tdm_load_tile64(ldsoff(&Bs[buf][0][0]),
                        Wcat + (long)n_blk*DMODEL + kb, 128u, (unsigned)DMODEL);
    };

    v8f acc[4][2] = {};
    auto compute = [&](int cur) {
#pragma unroll
        for (int kc = 0; kc < 2; ++kc) {  // two K=32 chunks of the K=64 tile
            v16bf a[4], b[2];
#pragma unroll
            for (int mt = 0; mt < 4; ++mt) {
                const __bf16* ap = &As[cur][wm + mt*16 + lc][kc*32 + half*8];
                a[mt] = pack16(*(const v8bf*)ap, *(const v8bf*)(ap + 16));
            }
#pragma unroll
            for (int nt = 0; nt < 2; ++nt) {
                const __bf16* bp = &Bs[cur][wn + nt*16 + lc][kc*32 + half*8];
                b[nt] = pack16(*(const v8bf*)bp, *(const v8bf*)(bp + 16));
            }
#pragma unroll
            for (int mt = 0; mt < 4; ++mt)
#pragma unroll
                for (int nt = 0; nt < 2; ++nt)
                    acc[mt][nt] = __builtin_amdgcn_wmma_f32_16x16x32_bf16(
                        false, a[mt], false, b[nt], (short)0, acc[mt][nt],
                        false, false);
        }
    };

    if (wave == 0) issue(0, 0);
    int cur = 0;
    for (int kb = 0; kb + 64 < DMODEL; kb += 64) {   // steady state, branch-free
        if (wave == 0) {
            issue(cur ^ 1, kb + 64);
            __builtin_amdgcn_s_wait_tensorcnt(2);    // cur's two tiles done
        }
        __syncthreads();
        compute(cur);
        __syncthreads();
        cur ^= 1;
    }
    if (wave == 0) __builtin_amdgcn_s_wait_tensorcnt(0);
    __syncthreads();
    compute(cur);

    const int sel = n_blk >> 10;  // 0=Q 1=K 2=V (uniform per block)
    const float* bias = (sel == 0) ? bq : (sel == 1) ? bk : bv;
#pragma unroll
    for (int mt = 0; mt < 4; ++mt) {
#pragma unroll
        for (int nt = 0; nt < 2; ++nt) {
            const int n  = n_blk + wn + nt*16 + lc;
            const int nl = n & (DMODEL - 1);
            const float bvv = bias[nl];
            const int h = nl >> 6, dh = nl & 63;
#pragma unroll
            for (int v = 0; v < 8; ++v) {
                const int m  = m_blk + wm + mt*16 + v + 8*half;
                const int b_ = m >> 11, s = m & (SEQ - 1);
                const __bf16 ov = (__bf16)(acc[mt][nt][v] + bvv);
                if (sel == 0)
                    Qh[((long)(b_*NHEAD + h)*SEQ + s)*DHEAD + dh] = ov;
                else if (sel == 1)
                    Kh[((long)(b_*NHEAD + h)*SEQ + s)*DHEAD + dh] = ov;
                else
                    Vt[((long)(b_*NHEAD + h)*DHEAD + dh)*SEQ + s] = ov;
            }
        }
    }
}

// ---------------------------------------------------------------------------
// Kernel 2: flash attention, ASYNCcnt-staged K/V shared by the block's
// 4 waves (same head). Wave = 16 queries, full Dh=64. Key chunks of 32:
// 2 score WMMAs x2 tiles -> online softmax -> P relayout via LDS -> 4 PV WMMAs.
// ---------------------------------------------------------------------------
__global__ __launch_bounds__(128) void attn_kernel(
    const __bf16* __restrict__ Qh, const __bf16* __restrict__ Kh,
    const __bf16* __restrict__ Vt,
    const float* __restrict__ mask,   // [B][S]
    float* __restrict__ out)          // [B,S,D]
{
    __shared__ __align__(16) __bf16 Ks[2][32][72];  // 32 keys x 64dh (pad->72)
    __shared__ __align__(16) __bf16 Vs[2][64][40];  // 64 dh x 32 keys (pad->40)
    __shared__ __align__(16) __bf16 Ps[4][16][32];  // per-wave P tile

    const int tid  = threadIdx.x;
    const int lane = tid & 31;
    const int wave = tid >> 5;
    const int half = lane >> 4;
    const int lc   = lane & 15;
    const int bh   = blockIdx.x;          // 0..63
    const int b    = bh >> 4, h = bh & 15;
    const int q0   = (blockIdx.y*4 + wave)*16;

    const __bf16* Qbase = Qh + (long)bh*SEQ*DHEAD;
    const __bf16* Kbase = Kh + (long)bh*SEQ*DHEAD;
    const __bf16* Vbase = Vt + (long)bh*DHEAD*SEQ;
    const float*  mrow  = mask + (long)b*SEQ;

    auto stage = [&](int buf, int kb) {   // 4 async b128 per thread (2 K + 2 V)
#pragma unroll
        for (int i = 0; i < 2; ++i) {
            const int lin = i*128 + tid;  // 0..255
            { const int r = lin >> 3, c = lin & 7;   // 32 rows x 8 chunks
              async_ld16(&Ks[buf][r][c*8], Kbase + (long)(kb + r)*DHEAD + c*8); }
            { const int r = lin >> 2, c = lin & 3;   // 64 rows x 4 chunks
              async_ld16(&Vs[buf][r][c*8], Vbase + (long)r*SEQ + kb + c*8); }
        }
    };

    // Q A-fragments (constant over the key loop)
    v16bf aq[2];
#pragma unroll
    for (int c = 0; c < 2; ++c) {
        const __bf16* qp = Qbase + (long)(q0 + lc)*DHEAD + c*32 + half*8;
        aq[c] = pack16(*(const v8bf*)qp, *(const v8bf*)(qp + 16));
    }

    v8f o[4] = {};
    float mrun[8], lsum[8];
#pragma unroll
    for (int v = 0; v < 8; ++v) { mrun[v] = -1e30f; lsum[v] = 0.0f; }

    auto body = [&](int cur, int kb) {
        float p0[8], p1[8];
#pragma unroll
        for (int tt = 0; tt < 2; ++tt) {
            v8f s = {};
#pragma unroll
            for (int c = 0; c < 2; ++c) {
                const __bf16* kp = &Ks[cur][tt*16 + lc][c*32 + half*8];
                v16bf bk16 = pack16(*(const v8bf*)kp, *(const v8bf*)(kp + 16));
                s = __builtin_amdgcn_wmma_f32_16x16x32_bf16(
                    false, aq[c], false, bk16, (short)0, s, false, false);
            }
            const float mval = mrow[kb + tt*16 + lc];
            float* dst = tt ? p1 : p0;
#pragma unroll
            for (int v = 0; v < 8; ++v) dst[v] = s[v]*0.125f + mval;
        }

        float alpha[8];
#pragma unroll
        for (int v = 0; v < 8; ++v) {
            float mt = fmaxf(p0[v], p1[v]);
#pragma unroll
            for (int off = 1; off < 16; off <<= 1)
                mt = fmaxf(mt, __shfl_xor(mt, off, 32));
            const float mnew = fmaxf(mrun[v], mt);
            const float e0 = __expf(p0[v] - mnew);
            const float e1 = __expf(p1[v] - mnew);
            float rs = e0 + e1;
#pragma unroll
            for (int off = 1; off < 16; off <<= 1)
                rs += __shfl_xor(rs, off, 32);
            const float a = __expf(mrun[v] - mnew);
            lsum[v] = lsum[v]*a + rs;
            mrun[v] = mnew;
            alpha[v] = a;
            const int r = v + 8*half;
            Ps[wave][r][lc]      = (__bf16)e0;
            Ps[wave][r][16 + lc] = (__bf16)e1;
        }
#pragma unroll
        for (int v = 0; v < 8; ++v) {
            o[0][v] *= alpha[v]; o[1][v] *= alpha[v];
            o[2][v] *= alpha[v]; o[3][v] *= alpha[v];
        }

        // P as A-fragment [16 q x 32 keys] (same-wave LDS ops are in-order)
        const __bf16* pp = &Ps[wave][lc][half*8];
        v16bf ap = pack16(*(const v8bf*)pp, *(const v8bf*)(pp + 16));
#pragma unroll
        for (int t = 0; t < 4; ++t) {
            const __bf16* vp = &Vs[cur][t*16 + lc][half*8];
            v16bf bv16 = pack16(*(const v8bf*)vp, *(const v8bf*)(vp + 16));
            o[t] = __builtin_amdgcn_wmma_f32_16x16x32_bf16(
                false, ap, false, bv16, (short)0, o[t], false, false);
        }
    };

    stage(0, 0);
    int cur = 0, kb = 0;
    for (; kb + 32 < SEQ; kb += 32) {                // branch-free steady state
        stage(cur ^ 1, kb + 32);
        wait_async_le4();
        __syncthreads();
        body(cur, kb);
        __syncthreads();
        cur ^= 1;
    }
    wait_async_0();
    __syncthreads();
    body(cur, kb);

    // Normalize and write ctx [B,S,D] fp32
#pragma unroll
    for (int t = 0; t < 4; ++t) {
        const int dh = t*16 + lc;
#pragma unroll
        for (int v = 0; v < 8; ++v) {
            const int s = q0 + v + 8*half;
            out[((long)b*SEQ + s)*DMODEL + h*DHEAD + dh] = o[t][v] / lsum[v];
        }
    }
}

// ---------------------------------------------------------------------------
extern "C" void kernel_launch(void* const* d_in, const int* in_sizes, int n_in,
                              void* d_out, int out_size, void* d_ws, size_t ws_size,
                              hipStream_t stream) {
    const float* hidden = (const float*)d_in[0];
    const float* mask   = (const float*)d_in[1];
    const float* Wq     = (const float*)d_in[2];
    const float* bq     = (const float*)d_in[3];
    const float* Wk     = (const float*)d_in[4];
    const float* bk     = (const float*)d_in[5];
    const float* Wv     = (const float*)d_in[6];
    const float* bv     = (const float*)d_in[7];
    float* out = (float*)d_out;

    char* ws = (char*)d_ws;
    __bf16* Xb   = (__bf16*)(ws);                    // 16 MB: [8192][1024]
    __bf16* Wcat = (__bf16*)(ws + (16L << 20));      //  6 MB: [3072][1024]
    __bf16* Qh   = (__bf16*)(ws + (22L << 20));      // 16 MB: [B,H,S,Dh]
    __bf16* Kh   = (__bf16*)(ws + (38L << 20));      // 16 MB: [B,H,S,Dh]
    __bf16* Vt   = (__bf16*)(ws + (54L << 20));      // 16 MB: [B,H,Dh,S]

    const long total4 = ((long)ROWS*DMODEL + (long)NQKV*DMODEL) / 4;
    cvt_kernel<<<(int)((total4 + 255) / 256), 256, 0, stream>>>(
        hidden, Wq, Wk, Wv, Xb, Wcat);

    dim3 g1(NQKV / 128, ROWS / 128);  // (24, 64)
    qkv_gemm_kernel<<<g1, 256, 0, stream>>>(Xb, Wcat, bq, bk, bv, Qh, Kh, Vt);

    dim3 g2(BATCH * NHEAD, SEQ / 64); // (64, 32)
    attn_kernel<<<g2, 128, 0, stream>>>(Qh, Kh, Vt, mask, out);
}
